// FALayer_20521353740421
// MI455X (gfx1250) — matrix operand order repaired and checked by reading, compiled, not run on hardware
//
#include <hip/hip_runtime.h>
#include <hip/hip_bf16.h>

// ---------------------------------------------------------------------------
// FALayer on MI455X (gfx1250, wave32, WMMA).
// Per-node projections (not per-edge), softmax collapsed to per-(row,head)
// scalars + sparse edge corrections, one dense A@X WMMA GEMM.
// GEMMs: fragment-packed f16 B operand; 32x64 C tile per wave (8 wmma/k-step).
// ---------------------------------------------------------------------------

#define NN   4096
#define DD   512
#define HH   8
#define EE   131072
#define DHH  64

typedef _Float16 v16h __attribute__((ext_vector_type(16)));
typedef float    v8f  __attribute__((ext_vector_type(8)));

// ---------------- conversion / packing kernels ----------------

__global__ void k_cvt_f16(const float* __restrict__ src, _Float16* __restrict__ dst, int n) {
    int t = blockIdx.x * 256 + threadIdx.x;
    if (t < n) dst[t] = (_Float16)src[t];
}

// Pack fp32 source directly into WMMA B-fragment layout (f16).
// Logical B is KxN. Tiles are 32x16 (K x N), 512 halves each, stored
// tile-contiguously; within a tile, lane L owns halves [L*16, L*16+16) with
// n = L&15, k_in_tile = (e&7) + (L>>4)*8 + (e>>3)*16  (CDNA5 ISA 7.12.2 16-bit B layout).
// transposeSrc==0 : B[k][n] = src[k*Ndim + n]   (src is KxN row-major)
// transposeSrc==1 : B[k][n] = src[n*K + k]      (src is NxK row-major, e.g. W)
__global__ void k_pack_b(const float* __restrict__ src, _Float16* __restrict__ Bp,
                         int K, int Ndim, int transposeSrc) {
    int t = blockIdx.x * 256 + threadIdx.x;           // t < K*Ndim
    int tile = t >> 9;                                // 512 halves per tile
    int idx  = t & 511;
    int lane = idx >> 4;
    int e    = idx & 15;
    int n    = lane & 15;
    int g    = lane >> 4;
    int tilesPerRow = Ndim >> 4;
    int kt = tile / tilesPerRow;
    int ct = tile - kt * tilesPerRow;
    int k   = kt * 32 + (e & 7) + g * 8 + (e >> 3) * 16;
    int col = ct * 16 + n;
    float v = transposeSrc ? src[(size_t)col * K + k] : src[(size_t)k * Ndim + col];
    Bp[t] = (_Float16)v;
}

// per-row nonzero count of aggregator + diagonal flag
__global__ void k_rowcount(const float* __restrict__ agg, float* validCnt, float* aggDiag) {
    int i = blockIdx.x, t = threadIdx.x;
    const float* row = agg + (size_t)i * NN;
    int c = 0;
    for (int j = t; j < NN; j += 256) c += (row[j] != 0.0f);
    c += __shfl_xor(c, 1); c += __shfl_xor(c, 2); c += __shfl_xor(c, 4);
    c += __shfl_xor(c, 8); c += __shfl_xor(c, 16);
    __shared__ int sh[8];
    if ((t & 31) == 0) sh[t >> 5] = c;
    __syncthreads();
    if (t == 0) {
        int s = 0;
        #pragma unroll
        for (int w = 0; w < 8; ++w) s += sh[w];
        validCnt[i] = (float)s;
        aggDiag[i]  = (row[i] != 0.0f) ? 1.0f : 0.0f;
    }
}

// ---------------- WMMA GEMM: C(MxN) = A(MxK,f16 row-major) * Bp(packed) + bias ----
// 8 waves/block arranged 2x4; each wave computes a 32x64 C tile:
// 2 A fragments x 4 B fragments -> 8 v_wmma per 32-deep k-step.
// Block tile 64x256. Grids are exact multiples -> no divergence (EXEC all-1s).

__global__ void k_wmma_gemm(const _Float16* __restrict__ A, const _Float16* __restrict__ Bp,
                            const float* __restrict__ bias, float* __restrict__ C,
                            int M, int Ndim, int K) {
    int lane = threadIdx.x & 31;
    int w    = threadIdx.x >> 5;
    int n    = lane & 15;
    int g    = lane >> 4;
    int r0   = blockIdx.y * 64 + (w >> 2) * 32;          // wave row base (2 row tiles)
    int c0   = blockIdx.x * 256 + (w & 3) * 64;          // wave col base (4 col tiles)
    int tilesPerRow = Ndim >> 4;
    int ctBase = c0 >> 4;

    v8f acc[2][4];
    #pragma unroll
    for (int j = 0; j < 4; ++j) {
        float bv = bias ? bias[c0 + j * 16 + n] : 0.0f;
        #pragma unroll
        for (int i = 0; i < 2; ++i)
            #pragma unroll
            for (int r = 0; r < 8; ++r) acc[i][j][r] = bv;
    }

    // A fragment base: lane owns row (r0 + i*16 + (lane&15)), halves at
    // k offsets g*8 + [0..7] and g*8 + [16..23]  -> two 16B loads each.
    const _Float16* aBase0 = A + (size_t)(r0 + n) * K + g * 8;
    const _Float16* aBase1 = A + (size_t)(r0 + 16 + n) * K + g * 8;

    for (int k0 = 0; k0 < K; k0 += 32) {
        __builtin_prefetch(aBase0 + k0 + 64, 0, 3);      // global_prefetch path
        v16h a[2], b[4];
        #pragma unroll
        for (int i = 0; i < 2; ++i) {
            const _Float16* pa = (i ? aBase1 : aBase0) + k0;
            #pragma unroll
            for (int v = 0; v < 8; ++v) {                // vectorizes to 2x b128
                int kk = (v < 4) ? (v * 2) : (16 + (v - 4) * 2);
                a[i][2 * v]     = pa[kk];
                a[i][2 * v + 1] = pa[kk + 1];
            }
        }
        size_t tileRow = (size_t)(k0 >> 5) * tilesPerRow;
        #pragma unroll
        for (int j = 0; j < 4; ++j)                      // packed: one 32B load each
            b[j] = *(const v16h*)(Bp + ((tileRow + ctBase + j) << 9) + lane * 16);
        #pragma unroll
        for (int i = 0; i < 2; ++i)
            #pragma unroll
            for (int j = 0; j < 4; ++j)
                acc[i][j] = __builtin_amdgcn_wmma_f32_16x16x32_f16(
                    false, a[i], false, b[j], (short)0, acc[i][j], false, false);
    }

    #pragma unroll
    for (int i = 0; i < 2; ++i)
        #pragma unroll
        for (int j = 0; j < 4; ++j)
            #pragma unroll
            for (int r = 0; r < 8; ++r)                  // C/D: M = r + 8*g, N = lane&15
                C[(size_t)(r0 + i * 16 + r + g * 8) * Ndim + c0 + j * 16 + n] = acc[i][j][r];
}

// ---------------- per-edge scores: one wave per edge ----------------

__global__ void k_scores(const float* __restrict__ Qf, const float* __restrict__ Kf,
                         const int* __restrict__ pairs, float* __restrict__ scores) {
    int wid  = (blockIdx.x * blockDim.x + threadIdx.x) >> 5;
    int lane = threadIdx.x & 31;
    if (wid >= EE) return;
    int sub = pairs[2 * wid], obj = pairs[2 * wid + 1];
    const float4* q = (const float4*)(Qf + (size_t)sub * DD + lane * 16);
    const float4* k = (const float4*)(Kf + (size_t)obj * DD + lane * 16);
    float acc = 0.0f;
    #pragma unroll
    for (int j = 0; j < 4; ++j) {
        float4 a = q[j], b = k[j];
        acc += a.x * b.x + a.y * b.y + a.z * b.z + a.w * b.w;
    }
    acc += __shfl_xor(acc, 1);                // 4 lanes -> one head (DH=64)
    acc += __shfl_xor(acc, 2);
    if ((lane & 3) == 0) scores[(size_t)wid * HH + (lane >> 2)] = acc * 0.125f;  // /sqrt(64)
}

// ---------------- CSR build ----------------

__global__ void k_init(int* cnt, int* cursor) {
    int t = blockIdx.x * 256 + threadIdx.x;
    if (t < NN) { cnt[t] = 0; cursor[t] = 0; }
}

__global__ void k_hist(const int* __restrict__ pairs, int* cnt) {
    int e = blockIdx.x * 256 + threadIdx.x;
    if (e < EE) atomicAdd(&cnt[pairs[2 * e]], 1);
}

__global__ void k_scan(const int* __restrict__ cnt, int* __restrict__ starts) {
    __shared__ int sh[256];
    int t = threadIdx.x, base = t * 16, local = 0;
    #pragma unroll
    for (int j = 0; j < 16; ++j) local += cnt[base + j];
    sh[t] = local;
    __syncthreads();
    for (int off = 1; off < 256; off <<= 1) {
        int v = (t >= off) ? sh[t - off] : 0;
        __syncthreads();
        sh[t] += v;
        __syncthreads();
    }
    int run = (t == 0) ? 0 : sh[t - 1];
    for (int j = 0; j < 16; ++j) { starts[base + j] = run; run += cnt[base + j]; }
    if (t == 255) starts[NN] = run;
}

__global__ void k_scatter(const int* __restrict__ pairs, const int* __restrict__ starts,
                          int* cursor, int* eidx) {
    int e = blockIdx.x * 256 + threadIdx.x;
    if (e >= EE) return;
    int r = pairs[2 * e];
    int p = starts[r] + atomicAdd(&cursor[r], 1);
    eidx[p] = e;
}

// ---------------- per-(row,head) softmax stats; 8 lanes per row ----------------

__global__ void k_rowstats(const int* __restrict__ starts, const int* __restrict__ eidx,
                           const int* __restrict__ pairs, const float* __restrict__ agg,
                           const float* __restrict__ scores,
                           const float* __restrict__ validCnt, const float* __restrict__ aggDiag,
                           float* __restrict__ mArr, float* __restrict__ dArr,
                           float* __restrict__ coefPlain, float* __restrict__ cDiag2) {
    int t = blockIdx.x * blockDim.x + threadIdx.x;
    int i = t >> 3, h = t & 7;
    if (i >= NN) return;
    int s0 = starts[i], s1 = starts[i + 1];
    float m = 1e-7f;           // diagonal always present with 1e-7 (> the 0-valued plain entries)
    int nValid = 0;
    for (int p = s0; p < s1; ++p) {
        int e = eidx[p], obj = pairs[2 * e + 1];
        if (obj == i || agg[(size_t)i * NN + obj] == 0.0f) continue;
        ++nValid;
        m = fmaxf(m, scores[(size_t)e * HH + h]);
    }
    float plain = validCnt[i] - aggDiag[i] - (float)nValid;   // zero-valued valid entries
    float denom = expf(1e-7f - m) + plain * expf(-m);
    for (int p = s0; p < s1; ++p) {
        int e = eidx[p], obj = pairs[2 * e + 1];
        if (obj == i || agg[(size_t)i * NN + obj] == 0.0f) continue;
        denom += expf(scores[(size_t)e * HH + h] - m);
    }
    mArr[t] = m; dArr[t] = denom;
    float pp = expf(-m) / denom;            // softmax weight of a plain (value 0) entry
    float dd = expf(1e-7f - m) / denom;     // softmax weight of the diagonal entry
    pp += __shfl_xor(pp, 1); pp += __shfl_xor(pp, 2); pp += __shfl_xor(pp, 4);
    dd += __shfl_xor(dd, 1); dd += __shfl_xor(dd, 2); dd += __shfl_xor(dd, 4);
    if (h == 0) {
        float cp = pp * 0.125f;
        coefPlain[i] = cp;
        cDiag2[i]    = dd * 0.125f - cp * aggDiag[i];
    }
}

// ---------------- per-edge combined softmax weight ----------------

__global__ void k_edgew(const int* __restrict__ pairs, const float* __restrict__ agg,
                        const float* __restrict__ scores, const float* __restrict__ mArr,
                        const float* __restrict__ dArr, float* __restrict__ dw) {
    int e = blockIdx.x * 256 + threadIdx.x;
    if (e >= EE) return;
    int i = pairs[2 * e], obj = pairs[2 * e + 1];
    float w = 0.0f;
    if (obj != i && agg[(size_t)i * NN + obj] != 0.0f) {
        #pragma unroll
        for (int h = 0; h < HH; ++h)
            w += expf(scores[(size_t)e * HH + h] - mArr[i * HH + h]) / dArr[i * HH + h];
    }
    dw[e] = w * 0.125f;
}

// ---------------- final combine: out[i] = cp*AX[i] + cD*x[i] + sum_e (dw-cp)*x[obj] --

__global__ void k_final(const float* __restrict__ X, const float* __restrict__ AX,
                        const float* __restrict__ agg, const int* __restrict__ starts,
                        const int* __restrict__ eidx, const int* __restrict__ pairs,
                        const float* __restrict__ dw, const float* __restrict__ coefPlain,
                        const float* __restrict__ cDiag2, float* __restrict__ out) {
    int i = blockIdx.x, t = threadIdx.x;
    float c0 = coefPlain[i], c1 = cDiag2[i];
    int s0 = starts[i], s1 = starts[i + 1];
    float a0 = c0 * AX[(size_t)i * DD + t]       + c1 * X[(size_t)i * DD + t];
    float a1 = c0 * AX[(size_t)i * DD + t + 256] + c1 * X[(size_t)i * DD + t + 256];
    for (int p = s0; p < s1; ++p) {
        int e = eidx[p], obj = pairs[2 * e + 1];
        if (obj == i || agg[(size_t)i * NN + obj] == 0.0f) continue;
        float w = dw[e] - c0;
        a0 += w * X[(size_t)obj * DD + t];
        a1 += w * X[(size_t)obj * DD + t + 256];
    }
    out[(size_t)i * DD + t]       = a0;
    out[(size_t)i * DD + t + 256] = a1;
}

// ---------------- host launch ----------------

extern "C" void kernel_launch(void* const* d_in, const int* in_sizes, int n_in,
                              void* d_out, int out_size, void* d_ws, size_t ws_size,
                              hipStream_t stream) {
    const float* X    = (const float*)d_in[0];
    const float* agg  = (const float*)d_in[1];
    const int*   prs  = (const int*)  d_in[2];
    const float* wq   = (const float*)d_in[3];
    const float* bq   = (const float*)d_in[4];
    const float* wk   = (const float*)d_in[5];
    const float* bk   = (const float*)d_in[6];
    float* out = (float*)d_out;

    char* p = (char*)d_ws;
    size_t off = 0;
    auto alloc = [&](size_t bytes) -> char* {
        char* r = p + off;
        off += (bytes + 255) & ~(size_t)255;
        return r;
    };
    _Float16* Xh   = (_Float16*)alloc((size_t)NN * DD * 2);   // A operand (row-major f16)
    _Float16* Ah   = (_Float16*)alloc((size_t)NN * NN * 2);   // A operand for A@X
    _Float16* BpWq = (_Float16*)alloc((size_t)DD * DD * 2);   // packed B operands
    _Float16* BpWk = (_Float16*)alloc((size_t)DD * DD * 2);
    _Float16* BpX  = (_Float16*)alloc((size_t)NN * DD * 2);
    float*    Qf   = (float*)   alloc((size_t)NN * DD * 4);
    float*    Kf   = (float*)   alloc((size_t)NN * DD * 4);
    float*    AX   = (float*)   alloc((size_t)NN * DD * 4);
    float*    sc   = (float*)   alloc((size_t)EE * HH * 4);
    float*    mArr = (float*)   alloc((size_t)NN * HH * 4);
    float*    dArr = (float*)   alloc((size_t)NN * HH * 4);
    float*    cfP  = (float*)   alloc((size_t)NN * 4);
    float*    cfD  = (float*)   alloc((size_t)NN * 4);
    float*    vCnt = (float*)   alloc((size_t)NN * 4);
    float*    aDg  = (float*)   alloc((size_t)NN * 4);
    int*      cnt  = (int*)     alloc((size_t)NN * 4);
    int*      curs = (int*)     alloc((size_t)NN * 4);
    int*      sts  = (int*)     alloc((size_t)(NN + 1) * 4);
    int*      eidx = (int*)     alloc((size_t)EE * 4);
    float*    dw   = (float*)   alloc((size_t)EE * 4);
    (void)ws_size; (void)in_sizes; (void)n_in; (void)out_size;

    // conversions / packing + aggregator row stats
    k_cvt_f16<<<(NN * DD) / 256, 256, 0, stream>>>(X, Xh, NN * DD);
    k_cvt_f16<<<(NN * NN) / 256, 256, 0, stream>>>(agg, Ah, NN * NN);
    k_pack_b <<<(DD * DD) / 256, 256, 0, stream>>>(wq, BpWq, DD, DD, 1);
    k_pack_b <<<(DD * DD) / 256, 256, 0, stream>>>(wk, BpWk, DD, DD, 1);
    k_pack_b <<<(NN * DD) / 256, 256, 0, stream>>>(X,  BpX,  NN, DD, 0);
    k_rowcount<<<NN, 256, 0, stream>>>(agg, vCnt, aDg);

    // WMMA GEMMs: block tile 64x256, wave tile 32x64
    dim3 blk(256);
    dim3 gP(DD / 256, NN / 64);      // projections: M=4096, N=512, K=512
    k_wmma_gemm<<<gP, blk, 0, stream>>>(Xh, BpWq, bq, Qf, NN, DD, DD);
    k_wmma_gemm<<<gP, blk, 0, stream>>>(Xh, BpWk, bk, Kf, NN, DD, DD);
    k_wmma_gemm<<<gP, blk, 0, stream>>>(Ah, BpX, nullptr, AX, NN, DD, NN);  // K=4096

    // CSR of edges by subject row
    k_init   <<<NN / 256, 256, 0, stream>>>(cnt, curs);
    k_hist   <<<EE / 256, 256, 0, stream>>>(prs, cnt);
    k_scan   <<<1, 256, 0, stream>>>(cnt, sts);
    k_scatter<<<EE / 256, 256, 0, stream>>>(prs, sts, curs, eidx);

    // scores, softmax stats, edge weights, final combine
    k_scores  <<<EE / 8, 256, 0, stream>>>(Qf, Kf, prs, sc);
    k_rowstats<<<(NN * HH) / 256, 256, 0, stream>>>(sts, eidx, prs, agg, sc, vCnt, aDg,
                                                    mArr, dArr, cfP, cfD);
    k_edgew   <<<EE / 256, 256, 0, stream>>>(prs, agg, sc, mArr, dArr, dw);
    k_final   <<<NN, 256, 0, stream>>>(X, AX, agg, sts, eidx, prs, dw, cfP, cfD, out);
}
